// GeometricAwareFeatureAggregator_16243566313721
// MI455X (gfx1250) — matrix operand order repaired and checked by reading, not compiled, not run on hardware
//
#include <hip/hip_runtime.h>
#include <math.h>
#include <stdint.h>

// ---------------- constants from the reference ----------------
#define BB   16
#define MM   128
#define NN   16384
#define DD   256
#define KNN  16
#define TAU  5.0f
#define BN_EPS 1e-5f
#define COS_EPS 1e-8f

typedef __attribute__((ext_vector_type(16))) _Float16 v16h;
typedef __attribute__((ext_vector_type(8)))  float    v8f;

// ---------------- async global->LDS staging (CDNA5 ASYNCcnt path) ----------
// Probe (round 2 stderr) showed the builtin's param 1 is
// 'int __vector(4) __device__ *' i.e. v4i in the global address space.
#if defined(__gfx1250__) && __has_builtin(__builtin_amdgcn_global_load_async_to_lds_b128) && __has_builtin(__builtin_amdgcn_s_wait_asynccnt)
#define USE_ASYNC 1
typedef int v4i_vs __attribute__((vector_size(16)));
typedef __attribute__((address_space(1))) v4i_vs g_v4i;
typedef __attribute__((address_space(3))) v4i_vs l_v4i;
__device__ __forceinline__ void async_copy16(const float* g, float* l) {
  __builtin_amdgcn_global_load_async_to_lds_b128(
      (g_v4i*)(uintptr_t)g, (l_v4i*)(uintptr_t)l, 0, 0);
}
__device__ __forceinline__ void async_wait() { __builtin_amdgcn_s_wait_asynccnt(0); }
#else
#define USE_ASYNC 0
__device__ __forceinline__ void async_wait() {}
#endif

// =====================================================================
// LDS-staged WMMA GEMM:  C[rows,256] = act( A1@W1^T (+ A2@W2^T) + bias (+res) )
// W layout: torch convention [256, ldw] row-major (k contiguous).
// Block = 256 threads (8 waves) computing a 64x64 output tile.
// Per 32-wide k-step: stage A(64x32 f32) and W(64x32 f32) tiles into LDS
// (async global->LDS when available), then each wave builds f16 fragments
// from LDS and issues v_wmma_f32_16x16x32_f16 into 2 accumulators.
// Grid: (rows/64, 256/64).
// =====================================================================
__global__ __launch_bounds__(256)
void gemm_wmma_kernel(const float* __restrict__ A1, int lda1,
                      const float* __restrict__ W1, int ldw1, int K1,
                      const float* __restrict__ A2, int lda2,
                      const float* __restrict__ W2, int ldw2, int K2,
                      const float* __restrict__ bias,
                      const float* __restrict__ res,
                      float* __restrict__ C, int act)
{
  __shared__ __align__(16) float sA[64 * 32];
  __shared__ __align__(16) float sW[64 * 32];

  const int t    = threadIdx.x;
  const int lane = t & 31;
  const int wv   = t >> 5;
  const int rsub = wv & 3;        // row subtile 0..3
  const int csel = wv >> 2;       // col subtile phase 0..1 (handles csel, csel+2)
  const int rowBase = blockIdx.x * 64;
  const int colBase = blockIdx.y * 64;

  const int lrow = rsub * 16 + (lane & 15);
  const int kbA  = ((lane >> 4) & 1) * 16;  // A half-row K base (0 or 16)
  const int kbB  = ((lane >> 4) & 1) * 16;  // B K base (0 or 16)

  v8f acc0 = {}, acc1 = {};

  const float* Ap = A1; int lda = lda1;
  const float* Wp = W1; int ldw = ldw1; int Kdim = K1;
  for (int pair = 0; pair < 2; ++pair) {
    for (int k0 = 0; k0 < Kdim; k0 += 32) {
      __syncthreads();                      // previous compute done with LDS
      // ---- stage tiles: 512 16B chunks per tile, 2 per thread per tile ----
      #pragma unroll
      for (int h = 0; h < 2; ++h) {
        const int q   = t + h * 256;
        const int rr  = q >> 3;
        const int seg = (q & 7) * 4;
        const float* ga = Ap + (size_t)(rowBase + rr) * lda + k0 + seg;
        const float* gw = Wp + (size_t)(colBase + rr) * ldw + k0 + seg;
#if USE_ASYNC
        async_copy16(ga, &sA[rr * 32 + seg]);
        async_copy16(gw, &sW[rr * 32 + seg]);
#else
        *(float4*)&sA[rr * 32 + seg] = *(const float4*)ga;
        *(float4*)&sW[rr * 32 + seg] = *(const float4*)gw;
#endif
        __builtin_prefetch(ga + 32, 0, 1);  // global_prefetch_b8 next k-step
        __builtin_prefetch(gw + 32, 0, 1);
      }
      async_wait();                         // wave's async loads landed in LDS
      __syncthreads();                      // all waves' tiles visible

      // ---- A fragment (shared by both col subtiles of this wave) ----
      v16h af;
      #pragma unroll
      for (int e = 0; e < 8; ++e) af[e]     = (_Float16)sA[lrow * 32 + kbA + e];
      #pragma unroll
      for (int e = 0; e < 8; ++e) af[8 + e] = (_Float16)sA[lrow * 32 + kbA + 8 + e];

      #pragma unroll
      for (int j = 0; j < 2; ++j) {
        const int lcol = (csel + 2 * j) * 16 + (lane & 15);
        v16h bf;
        #pragma unroll
        for (int e = 0; e < 16; ++e) bf[e] = (_Float16)sW[lcol * 32 + kbB + e];
        if (j == 0)
          acc0 = __builtin_amdgcn_wmma_f32_16x16x32_f16(false, af, false, bf,
                                                        (short)0, acc0, false, false);
        else
          acc1 = __builtin_amdgcn_wmma_f32_16x16x32_f16(false, af, false, bf,
                                                        (short)0, acc1, false, false);
      }
    }
    if (K2 <= 0) break;
    Ap = A2; lda = lda2; Wp = W2; ldw = ldw2; Kdim = K2;
  }

  // ---- epilogue: bias (+res) (+relu) ----
  const int hi = ((lane >> 4) & 1) * 8;
  #pragma unroll
  for (int j = 0; j < 2; ++j) {
    const int col = colBase + (csel + 2 * j) * 16 + (lane & 15);
    const float bv = bias ? bias[col] : 0.0f;
    v8f acc = j ? acc1 : acc0;
    #pragma unroll
    for (int v = 0; v < 8; ++v) {
      const int r = rowBase + rsub * 16 + hi + v;
      float x = acc[v] + bv;
      if (res) x += res[(size_t)r * DD + col];
      if (act) x = fmaxf(x, 0.0f);
      C[(size_t)r * DD + col] = x;
    }
  }
}

// =====================================================================
// KNN: one block per (b,m); 256 threads stream the 16384 points keeping
// per-thread sorted top-16, merge through LDS with 16 argmin rounds.
// Also emits knn_dxyz = k3[b,m] - pts[b,idx].
// =====================================================================
__global__ __launch_bounds__(256)
void knn_kernel(const float* __restrict__ pts, const float* __restrict__ k3,
                int* __restrict__ idx_out, float* __restrict__ knn_dxyz)
{
  const int bm = blockIdx.x;
  const int b  = bm >> 7;
  const int t  = threadIdx.x;
  __shared__ float sdist[256 * KNN];
  __shared__ int   sid  [256 * KNN];
  __shared__ float cd[256];
  __shared__ int   ct[256];

  const float kx = k3[bm * 3 + 0];
  const float ky = k3[bm * 3 + 1];
  const float kz = k3[bm * 3 + 2];
  const float* pb = pts + (size_t)b * NN * 3;

  float bd[KNN]; int bi[KNN];
  #pragma unroll
  for (int j = 0; j < KNN; ++j) { bd[j] = 3.0e38f; bi[j] = 0; }

  for (int p = t; p < NN; p += 256) {
    const float dx = kx - pb[p * 3 + 0];
    const float dy = ky - pb[p * 3 + 1];
    const float dz = kz - pb[p * 3 + 2];
    const float d2 = dx * dx + dy * dy + dz * dz;
    if (d2 < bd[KNN - 1]) {
      bd[KNN - 1] = d2; bi[KNN - 1] = p;
      #pragma unroll
      for (int j = KNN - 1; j > 0; --j) {
        if (bd[j] < bd[j - 1]) {
          float td = bd[j]; bd[j] = bd[j - 1]; bd[j - 1] = td;
          int   ti = bi[j]; bi[j] = bi[j - 1]; bi[j - 1] = ti;
        }
      }
    }
  }
  #pragma unroll
  for (int j = 0; j < KNN; ++j) { sdist[t * KNN + j] = bd[j]; sid[t * KNN + j] = bi[j]; }
  __syncthreads();

  int head = 0;
  for (int r = 0; r < KNN; ++r) {
    cd[t] = (head < KNN) ? sdist[t * KNN + head] : 3.0e38f;
    ct[t] = t;
    __syncthreads();
    for (int s = 128; s > 0; s >>= 1) {
      if (t < s && cd[t + s] < cd[t]) { cd[t] = cd[t + s]; ct[t] = ct[t + s]; }
      __syncthreads();
    }
    if (t == ct[0]) {
      const int p = sid[t * KNN + head];
      idx_out[bm * KNN + r] = p;
      knn_dxyz[(bm * KNN + r) * 3 + 0] = kx - pb[p * 3 + 0];
      knn_dxyz[(bm * KNN + r) * 3 + 1] = ky - pb[p * 3 + 1];
      knn_dxyz[(bm * KNN + r) * 3 + 2] = kz - pb[p * 3 + 2];
      head++;
    }
    __syncthreads();
  }
}

// gather: block per (b,m,r); 256 threads copy one 256-ch feature row
__global__ void gather_kernel(const float* __restrict__ ptsf,
                              const int* __restrict__ idx,
                              float* __restrict__ knn_fg)
{
  const int bmr = blockIdx.x;
  const int b = bmr >> 11;               // /(M*KNN) = /2048
  const int p = idx[bmr];
  knn_fg[(size_t)bmr * DD + threadIdx.x] =
      ptsf[((size_t)b * NN + p) * DD + threadIdx.x];
}

// Linear(3 -> 64) + ReLU (pos-MLP first layer)
__global__ void lin3_relu_kernel(const float* __restrict__ X,
                                 const float* __restrict__ W,
                                 const float* __restrict__ Bi,
                                 float* __restrict__ Y, int rows)
{
  const int i = blockIdx.x * blockDim.x + threadIdx.x;
  if (i >= rows * 64) return;
  const int row = i >> 6, n = i & 63;
  const float* x = X + row * 3;
  float y = Bi[n] + W[n * 3] * x[0] + W[n * 3 + 1] * x[1] + W[n * 3 + 2] * x[2];
  Y[i] = fmaxf(y, 0.0f);
}

// =====================================================================
// Fused dl branch: per (b,m) compute mean_j relu(L2(relu(L1(k3_m-k3_j))))
// L2 (64->256) via WMMA from LDS f16 tiles. 8 waves = 8 row tiles of j.
// (mean commutes with the affine dl3, applied afterwards as a 2048-row GEMM)
// =====================================================================
__global__ __launch_bounds__(256)
void dl_fused_kernel(const float* __restrict__ k3,
                     const float* __restrict__ w1, const float* __restrict__ b1,
                     const float* __restrict__ w2, const float* __restrict__ b2,
                     float* __restrict__ mean_h)
{
  const int bm = blockIdx.x;
  const int b = bm >> 7;
  const int tid = threadIdx.x;
  const int lane = tid & 31;
  const int wv = tid >> 5;

  __shared__ float    s_k3[MM * 3];
  __shared__ float    s_w1[64 * 3];
  __shared__ float    s_b1[64];
  __shared__ _Float16 s_h1[MM * 64];
  __shared__ _Float16 s_w2[DD * 64];
  __shared__ float    s_cs[DD];

  for (int i = tid; i < MM * 3; i += 256) s_k3[i] = k3[b * MM * 3 + i];
  for (int i = tid; i < 192; i += 256)    s_w1[i] = w1[i];
  if (tid < 64) s_b1[tid] = b1[tid];
  for (int i = tid; i < DD * 64; i += 256) s_w2[i] = (_Float16)w2[i];
  s_cs[tid] = 0.0f;
  const float mx = k3[bm * 3 + 0], my = k3[bm * 3 + 1], mz = k3[bm * 3 + 2];
  __syncthreads();

  for (int i = tid; i < MM * 64; i += 256) {
    const int j = i >> 6, cc = i & 63;
    const float dx = mx - s_k3[j * 3 + 0];
    const float dy = my - s_k3[j * 3 + 1];
    const float dz = mz - s_k3[j * 3 + 2];
    float h = s_w1[cc * 3] * dx + s_w1[cc * 3 + 1] * dy + s_w1[cc * 3 + 2] * dz + s_b1[cc];
    s_h1[i] = (_Float16)fmaxf(h, 0.0f);
  }
  __syncthreads();

  const int rrow = wv * 16 + (lane & 15);
  const int kbA = ((lane >> 4) & 1) * 16;
  const int kbB = ((lane >> 4) & 1) * 16;
  for (int ctile = 0; ctile < 16; ++ctile) {
    const int col = ctile * 16 + (lane & 15);
    v8f acc = {};
    #pragma unroll
    for (int ks = 0; ks < 2; ++ks) {
      v16h af, bf;
      const int ka = ks * 32 + kbA;
      #pragma unroll
      for (int e = 0; e < 8; ++e)  af[e]     = s_h1[rrow * 64 + ka + e];
      #pragma unroll
      for (int e = 0; e < 8; ++e)  af[8 + e] = s_h1[rrow * 64 + ka + 8 + e];
      const int kb = ks * 32 + kbB;
      #pragma unroll
      for (int e = 0; e < 16; ++e) bf[e]     = s_w2[col * 64 + kb + e];
      acc = __builtin_amdgcn_wmma_f32_16x16x32_f16(false, af, false, bf,
                                                   (short)0, acc, false, false);
    }
    const float bias = b2[col];
    float part = 0.0f;
    #pragma unroll
    for (int v = 0; v < 8; ++v) part += fmaxf(acc[v] + bias, 0.0f);
    atomicAdd(&s_cs[col], part);          // ds_add_f32
  }
  __syncthreads();
  mean_h[(size_t)bm * DD + tid] = s_cs[tid] * (1.0f / (float)MM);
}

// ------------------- BatchNorm (training-mode) -------------------
__global__ __launch_bounds__(256)
void bn_stats_kernel(const float* __restrict__ X, float* __restrict__ mean,
                     float* __restrict__ var)
{
  const int ch = blockIdx.x;
  const int t = threadIdx.x;
  __shared__ float rs[256], rs2[256];
  float s = 0.0f, s2 = 0.0f;
  for (int r = t; r < BB * MM; r += 256) {
    const float x = X[(size_t)r * DD + ch];
    s += x; s2 += x * x;
  }
  rs[t] = s; rs2[t] = s2; __syncthreads();
  for (int st = 128; st > 0; st >>= 1) {
    if (t < st) { rs[t] += rs[t + st]; rs2[t] += rs2[t + st]; }
    __syncthreads();
  }
  if (t == 0) {
    const float m = rs[0] / (float)(BB * MM);
    mean[ch] = m;
    var[ch] = rs2[0] / (float)(BB * MM) - m * m;
  }
}

__global__ void bn_apply_kernel(const float* __restrict__ X,
                                const float* __restrict__ g,
                                const float* __restrict__ bt,
                                const float* __restrict__ mean,
                                const float* __restrict__ var,
                                float* __restrict__ Y, int n, int relu)
{
  const int i = blockIdx.x * blockDim.x + threadIdx.x;
  if (i >= n) return;
  const int c = i & (DD - 1);
  float y = (X[i] - mean[c]) * rsqrtf(var[c] + BN_EPS) * g[c] + bt[c];
  if (relu) y = fmaxf(y, 0.0f);
  Y[i] = y;
}

// ------------------- cosine attention over k neighbors -------------------
__device__ __forceinline__ float bsum(float v, float* red)
{
  const int t = threadIdx.x;
  __syncthreads();
  red[t] = v; __syncthreads();
  for (int s = 128; s > 0; s >>= 1) {
    if (t < s) red[t] += red[t + s];
    __syncthreads();
  }
  return red[0];
}

__global__ __launch_bounds__(256)
void attn_kernel(const float* __restrict__ q, const float* __restrict__ knn3,
                 const float* __restrict__ pre, float* __restrict__ out)
{
  const int bm = blockIdx.x;
  const int c = threadIdx.x;
  __shared__ float red[256];
  __shared__ float s_sim[KNN];
  __shared__ float s_qn;

  const float qv = q[(size_t)bm * DD + c];
  const float qq = bsum(qv * qv, red);
  if (c == 0) s_qn = fmaxf(sqrtf(qq), COS_EPS);

  for (int r = 0; r < KNN; ++r) {
    const float kv = knn3[((size_t)bm * KNN + r) * DD + c];
    const float kk = bsum(kv * kv, red);
    const float qk = bsum(qv * kv, red);
    if (c == 0)
      s_sim[r] = qk / (s_qn * fmaxf(sqrtf(kk), COS_EPS)) * (1.0f / TAU);
  }
  __syncthreads();
  if (c == 0) {
    float mxv = -3.0e38f;
    for (int r = 0; r < KNN; ++r) mxv = fmaxf(mxv, s_sim[r]);
    float sum = 0.0f;
    for (int r = 0; r < KNN; ++r) { const float e = expf(s_sim[r] - mxv); s_sim[r] = e; sum += e; }
    for (int r = 0; r < KNN; ++r) s_sim[r] /= sum;
  }
  __syncthreads();
  float acc = 0.0f;
  for (int r = 0; r < KNN; ++r)
    acc += s_sim[r] * knn3[((size_t)bm * KNN + r) * DD + c];
  out[(size_t)bm * DD + c] = fmaxf(acc + pre[(size_t)bm * DD + c], 0.0f);
}

// ------------------- small helpers -------------------
__global__ void kg_mean_kernel(const float* __restrict__ kf, float* __restrict__ kg)
{
  const int b = blockIdx.x, c = threadIdx.x;
  float s = 0.0f;
  for (int m = 0; m < MM; ++m) s += kf[((size_t)b * MM + m) * DD + c];
  kg[b * DD + c] = s * (1.0f / (float)MM);
}

__global__ void concat2_kernel(const float* __restrict__ kg,
                               const float* __restrict__ posl,
                               float* __restrict__ cat2, int n)
{
  const int i = blockIdx.x * blockDim.x + threadIdx.x;
  if (i >= n) return;
  const int row = i >> 9, col = i & 511;
  const int b = row >> 7;
  cat2[i] = (col < DD) ? kg[b * DD + col]
                       : posl[(size_t)row * DD + (col - DD)];
}

__global__ void add_relu_kernel(const float* __restrict__ a,
                                const float* __restrict__ bptr,
                                float* __restrict__ out, int n)
{
  const int i = blockIdx.x * blockDim.x + threadIdx.x;
  if (i >= n) return;
  out[i] = fmaxf(a[i] + bptr[i], 0.0f);
}

// =====================================================================
// Host orchestration
// =====================================================================
extern "C" void kernel_launch(void* const* d_in, const int* in_sizes, int n_in,
                              void* d_out, int out_size, void* d_ws, size_t ws_size,
                              hipStream_t stream)
{
  (void)in_sizes; (void)n_in; (void)out_size; (void)ws_size;
  auto P = [&](int i) -> const float* { return (const float*)d_in[i]; };

  const float* kptf = P(0);   // [16,128,256]
  const float* k3   = P(1);   // [16,128,3]
  const float* ptsf = P(2);   // [16,16384,256]
  const float* pts  = P(3);   // [16,16384,3]

  // workspace carve-up (floats)
  float* wsf = (float*)d_ws;
  size_t off = 0;
  auto alloc = [&](size_t n) { float* p = wsf + off; off += n; return p; };
  float* big0   = alloc((size_t)BB * MM * KNN * DD);  // knn_fg / knn3
  float* big1   = alloc((size_t)BB * MM * KNN * DD);  // tmp
  float* big2   = alloc((size_t)BB * MM * KNN * DD);  // pos
  float* buf64  = alloc((size_t)BB * MM * KNN * 64);  // pos-MLP hidden (64ch)
  float* cat2   = alloc((size_t)BB * MM * 2 * DD);    // [kg, pos_l]
  float* bufA   = alloc((size_t)BB * MM * DD);
  float* bufB   = alloc((size_t)BB * MM * DD);
  float* bufC   = alloc((size_t)BB * MM * DD);
  float* kf_cur = alloc((size_t)BB * MM * DD);
  float* kg     = alloc((size_t)BB * DD);
  float* bnm    = alloc(DD);
  float* bnv    = alloc(DD);
  float* dxyz   = alloc((size_t)BB * MM * KNN * 3);
  int*   idx    = (int*)alloc((size_t)BB * MM * KNN);

  const int ROWS = BB * MM;            // 2048
  const int BROWS = BB * MM * KNN;     // 32768
  const int NELE = ROWS * DD;

  auto gemm = [&](const float* A1, int lda1, const float* W1, int ldw1, int K1,
                  const float* A2, int lda2, const float* W2, int ldw2, int K2,
                  const float* bias, const float* res, float* C,
                  int rows, int act) {
    dim3 g(rows / 64, DD / 64);
    gemm_wmma_kernel<<<g, 256, 0, stream>>>(A1, lda1, W1, ldw1, K1,
                                            A2, lda2, W2, ldw2, K2,
                                            bias, res, C, act);
  };
  auto blocks = [](int n) { return (n + 255) / 256; };

  // KNN + dxyz: block-invariant (same points, same k3, k=16 in both blocks)
  knn_kernel<<<ROWS, 256, 0, stream>>>(pts, k3, idx, dxyz);

  const float* kf_in = kptf;
  for (int i = 0; i < 2; ++i) {
    // per-block parameter slices (torch [out,in] weights, leading dim L=2)
    const float* in1w = P(4)  + i * DD * DD;   const float* in1b = P(5)  + i * DD;
    const float* bing = P(6)  + i * DD;        const float* binb = P(7)  + i * DD;
    const float* in2w = P(8)  + i * DD * DD;   const float* in2b = P(9)  + i * DD;
    const float* d1w  = P(10) + i * 64 * 3;    const float* d1b  = P(11) + i * 64;
    const float* d2w  = P(12) + i * DD * 64;   const float* d2b  = P(13) + i * DD;
    const float* d3w  = P(14) + i * DD * DD;   const float* d3b  = P(15) + i * DD;
    const float* dl1w = P(16) + i * 64 * 3;    const float* dl1b = P(17) + i * 64;
    const float* dl2w = P(18) + i * DD * 64;   const float* dl2b = P(19) + i * DD;
    const float* dl3w = P(20) + i * DD * DD;   const float* dl3b = P(21) + i * DD;
    const float* da1w = P(22) + i * 64 * 3;    const float* da1b = P(23) + i * 64;
    const float* da2w = P(24) + i * DD * 64;   const float* da2b = P(25) + i * DD;
    const float* da3w = P(26) + i * DD * DD;   const float* da3b = P(27) + i * DD;
    const float* dd1w = P(28) + i * DD * 2*DD; const float* dd1b = P(29) + i * DD;
    const float* dd2w = P(30) + i * DD * DD;   const float* dd2b = P(31) + i * DD;
    const float* f1w  = P(32) + i * DD * 3*DD; const float* f1b  = P(33) + i * DD;
    const float* b1g  = P(34) + i * DD;        const float* b1b  = P(35) + i * DD;
    const float* f2w  = P(36) + i * DD * DD;   const float* f2b  = P(37) + i * DD;
    const float* b2g  = P(38) + i * DD;        const float* b2b  = P(39) + i * DD;
    const float* o1w  = P(40) + i * DD * DD;   const float* o1b  = P(41) + i * DD;
    const float* o2w  = P(42) + i * DD * DD;   const float* o2b  = P(43) + i * DD;

    float* kf_out = (i == 0) ? kf_cur : (float*)d_out;

    // ---- q = in2( relu(bn(in1(kf))) ) ----
    gemm(kf_in, DD, in1w, DD, DD, nullptr,0,nullptr,0,0, in1b, nullptr, bufA, ROWS, 0);
    bn_stats_kernel<<<DD, 256, 0, stream>>>(bufA, bnm, bnv);
    bn_apply_kernel<<<blocks(NELE), 256, 0, stream>>>(bufA, bing, binb, bnm, bnv, bufB, NELE, 1);
    gemm(bufB, DD, in2w, DD, DD, nullptr,0,nullptr,0,0, in2b, nullptr, bufA, ROWS, 0);

    // ---- gather neighbor features ----
    gather_kernel<<<BROWS, DD, 0, stream>>>(ptsf, idx, big0);

    // ---- pos = d3( relu(d2( relu(d1(dxyz)) )) ) ----
    lin3_relu_kernel<<<blocks(BROWS * 64), 256, 0, stream>>>(dxyz, d1w, d1b, buf64, BROWS);
    gemm(buf64, 64, d2w, 64, 64, nullptr,0,nullptr,0,0, d2b, nullptr, big1, BROWS, 1);
    gemm(big1, DD, d3w, DD, DD, nullptr,0,nullptr,0,0, d3b, nullptr, big2, BROWS, 0);

    // ---- knn3 = dd2( relu(dd1([knn_f, pos])) )  (split-K, no concat buffer) ----
    gemm(big0, DD, dd1w, 2*DD, DD, big2, DD, dd1w + DD, 2*DD, DD, dd1b, nullptr, big1, BROWS, 1);
    gemm(big1, DD, dd2w, DD, DD, nullptr,0,nullptr,0,0, dd2b, nullptr, big0, BROWS, 0);

    // ---- cosine-sim attention + residual ----
    attn_kernel<<<ROWS, 256, 0, stream>>>(bufA, big0, kf_in, bufB);
    float* kf2 = bufB;  // post-attention features (pre for f-branch)

    // ---- pa = da-MLP(k3) ----
    lin3_relu_kernel<<<blocks(ROWS * 64), 256, 0, stream>>>(k3, da1w, da1b, buf64, ROWS);
    gemm(buf64, 64, da2w, 64, 64, nullptr,0,nullptr,0,0, da2b, nullptr, bufA, ROWS, 1);
    gemm(bufA, DD, da3w, DD, DD, nullptr,0,nullptr,0,0, da3b, nullptr, bufC, ROWS, 0); // pa

    // ---- pl: fused mean of dl2 hidden, then dl3 (mean commutes w/ affine) ----
    dl_fused_kernel<<<ROWS, 256, 0, stream>>>(k3, dl1w, dl1b, dl2w, dl2b, bufA);
    gemm(bufA, DD, dl3w, DD, DD, nullptr,0,nullptr,0,0, dl3b, bufC /*+pa*/, cat2 + ROWS*DD, ROWS, 0);
    float* posl = cat2 + (size_t)ROWS * DD;  // temporarily park pos_l

    // ---- kg + concat [kg, pos_l], then f1 as split-K GEMM ----
    kg_mean_kernel<<<BB, 256, 0, stream>>>(kf2, kg);
    concat2_kernel<<<blocks(ROWS * 2 * DD), 256, 0, stream>>>(kg, posl, cat2, ROWS * 2 * DD);
    gemm(kf2, DD, f1w, 3*DD, DD, cat2, 2*DD, f1w + DD, 3*DD, 2*DD, f1b, nullptr, bufA, ROWS, 0);
    bn_stats_kernel<<<DD, 256, 0, stream>>>(bufA, bnm, bnv);
    bn_apply_kernel<<<blocks(NELE), 256, 0, stream>>>(bufA, b1g, b1b, bnm, bnv, bufC, NELE, 1);
    gemm(bufC, DD, f2w, DD, DD, nullptr,0,nullptr,0,0, f2b, nullptr, bufA, ROWS, 0);
    bn_stats_kernel<<<DD, 256, 0, stream>>>(bufA, bnm, bnv);
    bn_apply_kernel<<<blocks(NELE), 256, 0, stream>>>(bufA, b2g, b2b, bnm, bnv, bufC, NELE, 1);
    add_relu_kernel<<<blocks(NELE), 256, 0, stream>>>(bufC, kf2, bufA, NELE);  // kf3 -> bufA

    // ---- output MLP + residual ----
    gemm(bufA, DD, o1w, DD, DD, nullptr,0,nullptr,0,0, o1b, nullptr, bufB, ROWS, 1);
    gemm(bufB, DD, o2w, DD, DD, nullptr,0,nullptr,0,0, o2b, nullptr, bufC, ROWS, 0);
    add_relu_kernel<<<blocks(NELE), 256, 0, stream>>>(bufA, bufC, kf_out, NELE);

    kf_in = kf_out;
  }
}
